// SparseMLP_16028817949060
// MI455X (gfx1250) — matrix operand order repaired and checked by reading, not compiled, not run on hardware
//
#include <hip/hip_runtime.h>
#include <hip/hip_bf16.h>

// ---------------------------------------------------------------------------
// SparseMLP on MI455X (gfx1250): 3x dense 4096^3 GEMM (+bias, +ReLU) chained.
// Compute-bound (4.1e11 FLOP vs ~270MB traffic => 11.6us @ 23.3TB/s), so we
// use the fastest dense matrix path: v_wmma_f32_16x16x32_bf16 (fp32 accum).
// Weights fit in the 192MB L2, so cross-block W-tile reuse is L2-resident.
//
// Round-2 change: double-buffered LDS + prefetch-distance-2 software pipeline
// -> one barrier per K-iteration, and each global load gets a full iteration
// of WMMA work to cover its latency (round-1 asm showed s_wait_loadcnt 0x0
// landing only ~4 WMMAs after the loads were issued).
// ---------------------------------------------------------------------------

typedef __attribute__((ext_vector_type(16))) __bf16 v16bf;
typedef __attribute__((ext_vector_type(8)))  __bf16 bf16x8;
typedef __attribute__((ext_vector_type(8)))  float  v8f;

#define NDIM       4096
#define TILE_M     128
#define TILE_N     128
#define TILE_K     32
#define LDS_STRIDE 40   // bf16 units: 80B rows -> conflict-free ds_load_b128
#define KITERS     (NDIM / TILE_K)

// Load a 16-element bf16 WMMA fragment from LDS as two b128 loads.
__device__ __forceinline__ v16bf ld_frag(const __bf16* p, int second_off) {
    bf16x8 lo = *reinterpret_cast<const bf16x8*>(p);
    bf16x8 hi = *reinterpret_cast<const bf16x8*>(p + second_off);
    return __builtin_shufflevector(lo, hi,
        0,1,2,3,4,5,6,7,8,9,10,11,12,13,14,15);
}

__device__ __forceinline__ bf16x8 cvt8(const float4 a, const float4 b) {
    bf16x8 r = { (__bf16)a.x, (__bf16)a.y, (__bf16)a.z, (__bf16)a.w,
                 (__bf16)b.x, (__bf16)b.y, (__bf16)b.z, (__bf16)b.w };
    return r;
}

// C[M][N] = act(A[M][K] @ W[N][K]^T + bias[N]);  M = N = K = 4096.
template<bool RELU>
__global__ __launch_bounds__(256)
void gemm_bias_act(const float* __restrict__ A,
                   const float* __restrict__ W,
                   const float* __restrict__ bias,
                   float* __restrict__ C)
{
    __shared__ __bf16 As[2][TILE_M * LDS_STRIDE];   // 2 x 10 KB
    __shared__ __bf16 Bs[2][TILE_N * LDS_STRIDE];   // 2 x 10 KB

    const int tid  = threadIdx.x;
    const int lane = tid & 31;          // wave32
    const int wave = tid >> 5;          // 0..7
    const int wm   = wave >> 1;         // 0..3 : 32 output rows each
    const int wn   = wave & 1;          // 0..1 : 64 output cols each
    const int lm   = lane & 15;
    const int lh   = lane >> 4;

    const int m0 = blockIdx.y * TILE_M;
    const int n0 = blockIdx.x * TILE_N;

    // --- staging map: 256 threads x 16 f32 = 128x32 tile -------------------
    const int srow  = tid >> 1;         // 0..127
    const int shalf = (tid & 1) * 16;   // 0 or 16 (fp32 columns)

    const float4* gA = reinterpret_cast<const float4*>(
        A + (size_t)(m0 + srow) * NDIM + shalf);
    const float4* gB = reinterpret_cast<const float4*>(
        W + (size_t)(n0 + srow) * NDIM + shalf);

    const int dstOff = srow * LDS_STRIDE + shalf;

    // --- per-wave fragment base offsets in LDS -----------------------------
    // A frag (16-bit 16x32 ISA layout): lane<16 holds K{0..7,16..23},
    // lane>=16 holds K{8..15,24..31}  -> base col = lh*8, second chunk +16.
    const int aOff = (wm * 32 + lm) * LDS_STRIDE + lh * 8;
    // B frag (32x16): same K order as A: lane<16 holds K{0..7,16..23},
    // lane>=16 holds K{8..15,24..31} -> base col = lh*8, second chunk +16.
    const int bOff = (wn * 64 + lm) * LDS_STRIDE + lh * 8;

    v8f acc[2][4];
    #pragma unroll
    for (int mi = 0; mi < 2; ++mi)
        #pragma unroll
        for (int ni = 0; ni < 4; ++ni) {
            v8f z = {0.f,0.f,0.f,0.f,0.f,0.f,0.f,0.f};
            acc[mi][ni] = z;
        }

    // --- prologue: tile 0 -> LDS buf 0, tile 1 -> registers ----------------
    float4 ra[4], rb[4];
    #pragma unroll
    for (int j = 0; j < 4; ++j) { ra[j] = gA[j]; rb[j] = gB[j]; }

    *reinterpret_cast<bf16x8*>(&As[0][dstOff])     = cvt8(ra[0], ra[1]);
    *reinterpret_cast<bf16x8*>(&As[0][dstOff + 8]) = cvt8(ra[2], ra[3]);
    *reinterpret_cast<bf16x8*>(&Bs[0][dstOff])     = cvt8(rb[0], rb[1]);
    *reinterpret_cast<bf16x8*>(&Bs[0][dstOff + 8]) = cvt8(rb[2], rb[3]);

    if (KITERS > 1) {
        #pragma unroll
        for (int j = 0; j < 4; ++j) { ra[j] = gA[8 + j]; rb[j] = gB[8 + j]; }
    }
    __syncthreads();

    // --- main loop: 1 barrier / iteration, prefetch distance 2 -------------
    for (int kt = 0; kt < KITERS; ++kt) {
        const int cur = kt & 1;
        const int nxt = cur ^ 1;

        // fragments + 8 WMMAs for this 32-deep K chunk (reads buf `cur`)
        v16bf af[2], bfr[4];
        #pragma unroll
        for (int mi = 0; mi < 2; ++mi)
            af[mi] = ld_frag(&As[cur][aOff + mi * 16 * LDS_STRIDE], 16);
        #pragma unroll
        for (int ni = 0; ni < 4; ++ni)
            bfr[ni] = ld_frag(&Bs[cur][bOff + ni * 16 * LDS_STRIDE], 16);

        #pragma unroll
        for (int mi = 0; mi < 2; ++mi)
            #pragma unroll
            for (int ni = 0; ni < 4; ++ni)
                acc[mi][ni] = __builtin_amdgcn_wmma_f32_16x16x32_bf16(
                    /*neg_a=*/false, af[mi],
                    /*neg_b=*/false, bfr[ni],
                    /*c_mod=*/(short)0, acc[mi][ni],
                    /*reuse_a=*/false, /*reuse_b=*/false);

        // stage tile kt+1 (already in regs) into buf `nxt`; safe: buf `nxt`
        // was last read in iteration kt-1, ordered by that barrier.
        if (kt + 1 < KITERS) {
            *reinterpret_cast<bf16x8*>(&As[nxt][dstOff])     = cvt8(ra[0], ra[1]);
            *reinterpret_cast<bf16x8*>(&As[nxt][dstOff + 8]) = cvt8(ra[2], ra[3]);
            *reinterpret_cast<bf16x8*>(&Bs[nxt][dstOff])     = cvt8(rb[0], rb[1]);
            *reinterpret_cast<bf16x8*>(&Bs[nxt][dstOff + 8]) = cvt8(rb[2], rb[3]);
        }

        // issue global loads for tile kt+2; a full iteration of WMMA work
        // (plus the barrier) now covers their latency.
        if (kt + 2 < KITERS) {
            #pragma unroll
            for (int j = 0; j < 4; ++j) {
                ra[j] = gA[(size_t)(kt + 2) * 8 + j];
                rb[j] = gB[(size_t)(kt + 2) * 8 + j];
            }
        }

        __syncthreads();
    }

    // --- epilogue: bias (+ReLU) and store ---------------------------------
    // C/D VGPR layout: lane<16 -> M = r, N = lane; lane>=16 -> M = r+8.
    #pragma unroll
    for (int ni = 0; ni < 4; ++ni) {
        const int gcol = n0 + wn * 64 + ni * 16 + lm;
        const float bv = bias[gcol];
        #pragma unroll
        for (int mi = 0; mi < 2; ++mi) {
            const int rbase = m0 + wm * 32 + mi * 16 + lh * 8;
            #pragma unroll
            for (int r = 0; r < 8; ++r) {
                float v = acc[mi][ni][r] + bv;
                if (RELU) v = fmaxf(v, 0.0f);
                C[(size_t)(rbase + r) * NDIM + gcol] = v;
            }
        }
    }
}

extern "C" void kernel_launch(void* const* d_in, const int* in_sizes, int n_in,
                              void* d_out, int out_size, void* d_ws, size_t ws_size,
                              hipStream_t stream) {
    const float* x  = (const float*)d_in[0];
    const float* W1 = (const float*)d_in[1];
    const float* b1 = (const float*)d_in[2];
    const float* W2 = (const float*)d_in[3];
    const float* b2 = (const float*)d_in[4];
    const float* W3 = (const float*)d_in[5];
    const float* b3 = (const float*)d_in[6];
    float* out = (float*)d_out;

    float* h1 = (float*)d_ws;                           // 4096*4096 f32
    float* h2 = h1 + (size_t)NDIM * NDIM;               // 4096*4096 f32

    dim3 grid(NDIM / TILE_N, NDIM / TILE_M);            // 32 x 32
    dim3 block(256);

    gemm_bias_act<true ><<<grid, block, 0, stream>>>(x,  W1, b1, h1);
    gemm_bias_act<true ><<<grid, block, 0, stream>>>(h1, W2, b2, h2);
    gemm_bias_act<false><<<grid, block, 0, stream>>>(h2, W3, b3, out);
}
